// EquivariantCorrectionHead_87445534147421
// MI455X (gfx1250) — matrix operand order repaired and checked by reading, not compiled, not run on hardware
//
#include <hip/hip_runtime.h>
#include <hip/hip_bf16.h>

// ---------------------------------------------------------------------------
// EquivariantCorrectionHead for MI455X (gfx1250, wave32, WMMA).
//
// B=131072, NS=64, H=32.  Strategy:
//   K1a: pre-swizzle big stage-1 weight [64 x 2208] f16 into WMMA-B tile layout
//        N = [2048 sss (v,w) | 160 combined stt+tst (j,w)]
//   K1b: pre-swizzle 4 small 32x32 f16 B-matrices (tts/ttt pair-sym, cd=stt+tst^T, w2_ttt)
//   K1c: build Wigner C222 (125 f32)
//   K2 : WMMA GEMM S(16x64) x Wcat(64x2208) with Wcat resident in 276KB LDS
//        (filled via GLOBAL_LOAD_ASYNC_TO_LDS_B128 when available),
//        fused reduction h_s0[b,w] = sum_v Y[b,v,w] s[b,v]; emits h_s0 (f32) + coef (f16)
//   K3 : per-16-row fused finish: per-element invariants (g, Q) -> WMMA (tts/ttt),
//        h_s/h_t finalize -> WMMA tp2 (cd, P) -> VALU epilogue -> out (B x 5 f32)
//
// Workspace requirement: ~59 MB (Wcat 276KB + small mats + C + hs0 16MB + coef 40MB).
// ---------------------------------------------------------------------------

typedef __attribute__((ext_vector_type(16))) _Float16 v16h;
typedef __attribute__((ext_vector_type(8)))  float    v8f;
typedef int v4i_ __attribute__((vector_size(16)));

union V16 { v16h h; int4 i2[2]; };
union H8  { int4 i; _Float16 h[8]; };
union HP  { int i; _Float16 h[2]; };

#define B_TOT    131072
#define NT1      138                 // stage-1 N tiles: 128 sss + 10 coef
#define WCAT_DW  (NT1 * 2 * 32 * 8)  // 70656 dwords of packed f16 pairs
#define WCAT_I4  (WCAT_DW / 4)       // 17664 int4

#define PW1_0   0.01557754f   // (NS^2+25)^-0.5
#define PW1_2   0.08671100f   // sqrt(5/665)
#define PW2_2   0.04034357f   // sqrt(5/3072)
#define INV_S5  0.44721360f

// CDNA5 async global->LDS copy (ASYNCcnt-tracked), guarded so absence of the
// builtin on a given toolchain falls back to the manual copy path.
#if defined(__has_builtin)
#  if __has_builtin(__builtin_amdgcn_global_load_async_to_lds_b128)
#    define ASYNC_LDS_COPY 1
#  endif
#endif

__device__ __forceinline__ void wait_asynccnt0() {
#if defined(__has_builtin) && __has_builtin(__builtin_amdgcn_s_wait_asynccnt)
    __builtin_amdgcn_s_wait_asynccnt(0);
#else
    asm volatile("s_wait_asynccnt 0x0" ::: "memory");
#endif
}

__device__ __forceinline__ v8f wmma32(v16h a, v16h b, v8f c) {
    return __builtin_amdgcn_wmma_f32_16x16x32_f16(false, a, false, b, (short)0, c, false, false);
}
__device__ __forceinline__ v16h ldA16(const _Float16* row, int hi8) {
    V16 u; u.i2[0] = *(const int4*)(row + 2 * hi8); u.i2[1] = *(const int4*)(row + 2 * hi8 + 8);
    return u.h;
}
__device__ __forceinline__ v16h ldB16(const int4* p) {
    V16 u; u.i2[0] = p[0]; u.i2[1] = p[1]; return u.h;
}

// upper-triangle pair tables for the symmetric (u<=v) 2e x 2e paths
static __device__ const int PU[15] = {0,0,0,0,0,1,1,1,1,2,2,2,3,3,4};
static __device__ const int PV[15] = {0,1,2,3,4,1,2,3,4,2,3,4,3,4,4};

// ---------------------------------------------------------------------------
// K1a: stage-1 big B matrix, f16, pre-swizzled into WMMA B-tile layout.
// Layout: dword d = ((nt*2+kb)*32 + lane)*8 + r ; value pair (K = kb*32 +
// (lane>=16?16:0) + 2r + {0,1}, N = nt*16 + (lane&15)).
// ---------------------------------------------------------------------------
__global__ void k_prep_wcat(const float* __restrict__ w_sss,
                            const float* __restrict__ w_stt,
                            const float* __restrict__ w_tst,
                            int* __restrict__ wcat) {
    int d = blockIdx.x * 256 + threadIdx.x;
    if (d >= WCAT_DW) return;
    int r     = d & 7;
    int lane  = (d >> 3) & 31;
    int kb    = (d >> 8) & 1;
    int nt    = d >> 9;
    int n     = nt * 16 + (lane & 15);
    int kbase = kb * 32 + ((lane >> 4) << 4) + 2 * r;
    HP p;
#pragma unroll
    for (int h = 0; h < 2; ++h) {
        int k = kbase + h;            // u (scalar) index 0..63
        float x;
        if (n < 2048) {               // sss block: n = v*32 + w
            int v = n >> 5, w = n & 31;
            x = w_sss[(k * 64 + v) * 32 + w];
        } else {                      // combined stt/tst block: c = j*32 + w
            int c = n - 2048; int j = c >> 5, w = c & 31;
            x = w_stt[(k * 5 + j) * 32 + w] + w_tst[(j * 64 + k) * 32 + w];
        }
        p.h[h] = (_Float16)x;
    }
    wcat[d] = p.i;
}

// ---------------------------------------------------------------------------
// K1b: four small 32x32 f16 B-matrices (K=32 single-WMMA), pre-swizzled.
//   mi=0: Wtts  [p,w]  = INV_S5 * sym(w1_tts)   (p<15, else 0)
//   mi=1: Wttt  [p,w]  = sym(w1_ttt)            (p<15, else 0)
//   mi=2: Wcd   [u,v]  = INV_S5 * (w2_stt[u,v] + w2_tst[v,u])
//   mi=3: W2t   [u,v]  = w2_ttt[u,v]
// ---------------------------------------------------------------------------
__global__ void k_prep_small(const float* __restrict__ w1_tts,
                             const float* __restrict__ w1_ttt,
                             const float* __restrict__ w2_stt,
                             const float* __restrict__ w2_tst,
                             const float* __restrict__ w2_ttt,
                             int* __restrict__ wsm) {
    int d = blockIdx.x * 256 + threadIdx.x;
    if (d >= 4 * 512) return;
    int mi = d >> 9, e = d & 511;
    int r = e & 7, lane = (e >> 3) & 31, ntile = (e >> 8) & 1;
    int n  = ntile * 16 + (lane & 15);
    int k0 = ((lane >> 4) << 4) + 2 * r;
    HP p;
#pragma unroll
    for (int h = 0; h < 2; ++h) {
        int k = k0 + h;
        float x = 0.0f;
        if (mi == 0) {
            if (k < 15) {
                int u = PU[k], v = PV[k];
                x = (u == v) ? w1_tts[(u * 5 + v) * 32 + n]
                             : (w1_tts[(u * 5 + v) * 32 + n] + w1_tts[(v * 5 + u) * 32 + n]);
                x *= INV_S5;
            }
        } else if (mi == 1) {
            if (k < 15) {
                int u = PU[k], v = PV[k];
                x = (u == v) ? w1_ttt[(u * 5 + v) * 32 + n]
                             : (w1_ttt[(u * 5 + v) * 32 + n] + w1_ttt[(v * 5 + u) * 32 + n]);
            }
        } else if (mi == 2) {
            x = INV_S5 * (w2_stt[k * 32 + n] + w2_tst[n * 32 + k]);
        } else {
            x = w2_ttt[k * 32 + n];
        }
        p.h[h] = (_Float16)x;
    }
    wsm[d] = p.i;
}

// ---------------------------------------------------------------------------
// K1c: real Wigner-3j(2,2,2) invariant tensor, Frobenius-normalized.
// ---------------------------------------------------------------------------
__global__ void k_prep_c222(float* __restrict__ C) {
    if (threadIdx.x != 0 || blockIdx.x != 0) return;
    float M[5][3][3];
    for (int a = 0; a < 5; ++a)
        for (int i = 0; i < 3; ++i)
            for (int j = 0; j < 3; ++j) M[a][i][j] = 0.0f;
    const float s2  = 0.70710678f;   // 1/sqrt(2)
    const float s6i = 0.40824829f;   // 1/sqrt(6)
    M[0][0][1] = s2;  M[0][1][0] = s2;
    M[1][1][2] = s2;  M[1][2][1] = s2;
    M[2][0][0] = -s6i; M[2][1][1] = -s6i; M[2][2][2] = 2.0f * s6i;
    M[3][0][2] = s2;  M[3][2][0] = s2;
    M[4][0][0] = s2;  M[4][1][1] = -s2;
    float T[125];
    for (int a = 0; a < 5; ++a)
        for (int b = 0; b < 5; ++b)
            for (int c = 0; c < 5; ++c) {
                float s = 0.0f;
                for (int i = 0; i < 3; ++i)
                    for (int j = 0; j < 3; ++j)
                        for (int k = 0; k < 3; ++k)
                            s += M[a][i][j] * M[b][j][k] * M[c][k][i];
                T[(a * 5 + b) * 5 + c] = s;
            }
    float S[125]; float nrm = 0.0f;
    for (int a = 0; a < 5; ++a)
        for (int b = 0; b < 5; ++b)
            for (int c = 0; c < 5; ++c) {
                float v = 0.5f * (T[(a * 5 + b) * 5 + c] + T[(b * 5 + a) * 5 + c]);
                S[(a * 5 + b) * 5 + c] = v; nrm += v * v;
            }
    nrm = sqrtf(nrm);
    for (int z = 0; z < 125; ++z) C[z] = S[z] / nrm;
}

// ---------------------------------------------------------------------------
// K2: stage-1 GEMM + fused sss reduction. 256 threads = 8 waves; Wcat lives
// in 276KB LDS shared by all waves; each wave streams 4 M-tiles of 16 rows.
// ---------------------------------------------------------------------------
__global__ __launch_bounds__(256) void k_stage1(
    const float*    __restrict__ sg,      // (B,64) f32
    const int4*     __restrict__ wcatg,   // WCAT_I4
    float*          __restrict__ hs0g,    // (B,32) f32
    _Float16*       __restrict__ coefg)   // (B,5,32) f16
{
    __shared__ __align__(16) int4     wcat4[WCAT_I4];       // 282624 B
    __shared__ __align__(16) _Float16 s16 [8][16][64];      //  16384 B
    __shared__ __align__(16) _Float16 sT16[8][64][16];      //  16384 B

    const int tid = threadIdx.x, wv = tid >> 5, lane = tid & 31;

#ifdef ASYNC_LDS_COPY
    // CDNA5 path: global -> LDS without touching VGPRs; ASYNCcnt-tracked.
    for (int i = tid; i < WCAT_I4; i += 256) {
        __builtin_amdgcn_global_load_async_to_lds_b128(
            (__attribute__((address_space(1))) v4i_*)(wcatg + i),
            (__attribute__((address_space(3))) v4i_*)(&wcat4[i]),
            0, 0);
    }
    wait_asynccnt0();
#else
    for (int i = tid; i < WCAT_I4; i += 256) wcat4[i] = wcatg[i];
#endif
    __syncthreads();

    const int m  = lane & 15;
    const int hi = lane >> 4;
    const int hi8 = hi * 8;

    for (int it = 0; it < 4; ++it) {
        const int tile = (blockIdx.x * 8 + wv) * 4 + it;
        const int b0 = tile * 16;

        // prefetch next tile's s row (lowers to global_prefetch_b8)
        if (it < 3)
            __builtin_prefetch(sg + (size_t)(b0 + 16 + m) * 64 + hi * 32, 0, 1);

        // stage s tile: row-major f16 (WMMA-A source) + transposed f16 (reduce source)
        {
            const int c0 = hi * 32;
            const float4* srcv = (const float4*)(sg + (size_t)(b0 + m) * 64 + c0);
#pragma unroll
            for (int q = 0; q < 8; ++q) {
                float4 f = srcv[q];
                int c = c0 + q * 4;
                HP a, b;
                a.h[0] = (_Float16)f.x; a.h[1] = (_Float16)f.y;
                b.h[0] = (_Float16)f.z; b.h[1] = (_Float16)f.w;
                *(int*)&s16[wv][m][c]     = a.i;
                *(int*)&s16[wv][m][c + 2] = b.i;
                sT16[wv][c + 0][m] = a.h[0]; sT16[wv][c + 1][m] = a.h[1];
                sT16[wv][c + 2][m] = b.h[0]; sT16[wv][c + 3][m] = b.h[1];
            }
        }
        asm volatile("s_wait_dscnt 0x0" ::: "memory");

        const v16h a0 = ldA16(&s16[wv][m][0],  hi8);       // K 0..31
        const v16h a1 = ldA16(&s16[wv][m][32], hi8);       // K 32..63

        float hsl[8], hsh[8];
#pragma unroll
        for (int r = 0; r < 8; ++r) { hsl[r] = 0.0f; hsh[r] = 0.0f; }

        // sss N-tiles: pair (2v, 2v+1) shares the s column v
        for (int v = 0; v < 64; ++v) {
            v8f acc0 = {0.f,0.f,0.f,0.f,0.f,0.f,0.f,0.f};
            v8f acc1 = {0.f,0.f,0.f,0.f,0.f,0.f,0.f,0.f};
            int base0 = ((2 * v) * 2) * 64 + lane * 2;     // int4 units
            acc0 = wmma32(a0, ldB16(&wcat4[base0]),      acc0);
            acc0 = wmma32(a1, ldB16(&wcat4[base0 + 64]), acc0);
            int base1 = ((2 * v + 1) * 2) * 64 + lane * 2;
            acc1 = wmma32(a0, ldB16(&wcat4[base1]),      acc1);
            acc1 = wmma32(a1, ldB16(&wcat4[base1 + 64]), acc1);
            H8 sh; sh.i = *(const int4*)&sT16[wv][v][hi8]; // broadcast read
#pragma unroll
            for (int r = 0; r < 8; ++r) {
                float sf = (float)sh.h[r];
                hsl[r] += acc0[r] * sf;    // w = m
                hsh[r] += acc1[r] * sf;    // w = 16 + m
            }
        }
        // coef N-tiles (combined stt+tst): store raw GEMM output
        for (int c2 = 0; c2 < 10; ++c2) {
            int nt = 128 + c2;
            v8f acc = {0.f,0.f,0.f,0.f,0.f,0.f,0.f,0.f};
            int base = (nt * 2) * 64 + lane * 2;
            acc = wmma32(a0, ldB16(&wcat4[base]),      acc);
            acc = wmma32(a1, ldB16(&wcat4[base + 64]), acc);
            const int j = c2 >> 1, w = (c2 & 1) * 16 + m;
#pragma unroll
            for (int r = 0; r < 8; ++r) {
                int b = b0 + hi8 + r;
                coefg[((size_t)b * 5 + j) * 32 + w] = (_Float16)acc[r];
            }
        }
#pragma unroll
        for (int r = 0; r < 8; ++r) {
            int b = b0 + hi8 + r;
            hs0g[(size_t)b * 32 + m]      = hsl[r];
            hs0g[(size_t)b * 32 + 16 + m] = hsh[r];
        }
    }
}

// ---------------------------------------------------------------------------
// K3: fused finish.  One wave per 16 batch rows; lane pairs (2 lanes / row)
// split the w / v halves.  All staging is wave-private LDS (in-order DS).
// ---------------------------------------------------------------------------
__global__ __launch_bounds__(256) void k_stage2(
    const float* __restrict__ t0, const float* __restrict__ t1,
    const float* __restrict__ t2, const float* __restrict__ t3,
    const float* __restrict__ t4,
    const float*    __restrict__ hs0g,
    const _Float16* __restrict__ coefg,
    const int4*     __restrict__ wsm4,    // 4 mats x 128 int4
    const float*    __restrict__ c222,
    float*          __restrict__ outg)
{
    __shared__ float cls[128];
    __shared__ __align__(16) float    tls [8][16][25];   // 12.8 KB
    __shared__ __align__(16) _Float16 A1  [8][96][32];   // 48   KB (later f32 scratch)
    __shared__ __align__(16) float    cres[8][96][32];   // 96   KB
    __shared__ __align__(16) _Float16 hstg[8][16][32];   //  8   KB
    __shared__ __align__(16) _Float16 htA [8][80][32];   // 40   KB

    const int tid = threadIdx.x, wv = tid >> 5, lane = tid & 31;
    if (tid < 125) cls[tid] = c222[tid];
    __syncthreads();

    const int tile = blockIdx.x * 8 + wv;
    const int b0 = tile * 16;
    const int m = lane & 15, hi = lane >> 4, hi8 = hi * 8;
    const int bl = lane >> 1, half = lane & 1;

    // prefetch the per-row global streams we consume in later phases
    __builtin_prefetch(hs0g + (size_t)(b0 + bl) * 32, 0, 1);
    __builtin_prefetch(coefg + (size_t)(b0 + bl) * 160, 0, 1);

    // ---- load t tile -> LDS ----
    for (int f = lane; f < 400; f += 32) {
        int mm = f / 25, rem = f % 25;
        int u = rem / 5, i = rem % 5;
        size_t gidx = (size_t)(b0 + mm) * 5 + i;
        float x = (u == 0) ? t0[gidx] : (u == 1) ? t1[gidx] :
                  (u == 2) ? t2[gidx] : (u == 3) ? t3[gidx] : t4[gidx];
        tls[wv][mm][rem] = x;
    }
    asm volatile("s_wait_dscnt 0x0" ::: "memory");

    float t[25];
#pragma unroll
    for (int z = 0; z < 25; ++z) t[z] = tls[wv][bl][z];

    // ---- zero-pad A1 rows for this b (g row + 5 Q rows), halves split dwords ----
#pragma unroll
    for (int rr = 0; rr < 6; ++rr) {
        int row = (rr == 0) ? bl : (16 + bl * 5 + (rr - 1));
        int* rp = (int*)&A1[wv][row][0];
#pragma unroll
        for (int q = 0; q < 8; ++q) rp[half * 8 + q] = 0;
    }
    // ---- invariants g[p], Q[p][k]  (both pair lanes write same values) ----
#pragma unroll
    for (int u = 0; u < 5; ++u) {
        float R[25];
#pragma unroll
        for (int j = 0; j < 5; ++j)
#pragma unroll
            for (int k = 0; k < 5; ++k) {
                float s = 0.0f;
#pragma unroll
                for (int i = 0; i < 5; ++i) s += cls[(i * 5 + j) * 5 + k] * t[u * 5 + i];
                R[j * 5 + k] = s;
            }
#pragma unroll
        for (int v = u; v < 5; ++v) {
            int p = 5 * u - (u * (u - 1)) / 2 + (v - u);
            float gp = 0.0f;
#pragma unroll
            for (int i = 0; i < 5; ++i) gp += t[u * 5 + i] * t[v * 5 + i];
            A1[wv][bl][p] = (_Float16)gp;
#pragma unroll
            for (int k = 0; k < 5; ++k) {
                float q = 0.0f;
#pragma unroll
                for (int j = 0; j < 5; ++j) q += R[j * 5 + k] * t[v * 5 + j];
                A1[wv][16 + bl * 5 + k][p] = (_Float16)q;
            }
        }
    }
    asm volatile("s_wait_dscnt 0x0" ::: "memory");

    // ---- hidden-layer WMMAs: tts (rows 0..15), ttt (rows 16..95) ----
    {
        v16h aT = ldA16(&A1[wv][m][0], hi8);
#pragma unroll
        for (int nt = 0; nt < 2; ++nt) {
            v8f c = {0.f,0.f,0.f,0.f,0.f,0.f,0.f,0.f};
            c = wmma32(aT, ldB16(&wsm4[0 * 128 + (nt * 32 + lane) * 2]), c);
#pragma unroll
            for (int r = 0; r < 8; ++r) cres[wv][hi8 + r][nt * 16 + m] = c[r];
        }
#pragma unroll
        for (int mt = 0; mt < 5; ++mt) {
            v16h aQ = ldA16(&A1[wv][16 + mt * 16 + m][0], hi8);
#pragma unroll
            for (int nt = 0; nt < 2; ++nt) {
                v8f c = {0.f,0.f,0.f,0.f,0.f,0.f,0.f,0.f};
                c = wmma32(aQ, ldB16(&wsm4[1 * 128 + (nt * 32 + lane) * 2]), c);
#pragma unroll
                for (int r = 0; r < 8; ++r) cres[wv][16 + mt * 16 + hi8 + r][nt * 16 + m] = c[r];
            }
        }
    }
    asm volatile("s_wait_dscnt 0x0" ::: "memory");

    // ---- finalize h_s / h_t (lane pair splits w halves) ----
    {
        const float4* hr = (const float4*)(hs0g + (size_t)(b0 + bl) * 32 + half * 16);
        float hs0v[16];
#pragma unroll
        for (int q = 0; q < 4; ++q) {
            float4 f = hr[q];
            hs0v[q * 4] = f.x; hs0v[q * 4 + 1] = f.y; hs0v[q * 4 + 2] = f.z; hs0v[q * 4 + 3] = f.w;
        }
#pragma unroll
        for (int q = 0; q < 8; ++q) {
            int w0 = half * 16 + 2 * q;
            HP pk2;
            pk2.h[0] = (_Float16)(PW1_0 * (hs0v[2 * q]     + cres[wv][bl][w0]));
            pk2.h[1] = (_Float16)(PW1_0 * (hs0v[2 * q + 1] + cres[wv][bl][w0 + 1]));
            *(int*)&hstg[wv][bl][w0] = pk2.i;
        }
        float cf[5][16];
#pragma unroll
        for (int j = 0; j < 5; ++j) {
            const int4* cp = (const int4*)(coefg + ((size_t)(b0 + bl) * 5 + j) * 32 + half * 16);
            H8 x0, x1; x0.i = cp[0]; x1.i = cp[1];
#pragma unroll
            for (int z = 0; z < 8; ++z) { cf[j][z] = (float)x0.h[z]; cf[j][8 + z] = (float)x1.h[z]; }
        }
#pragma unroll
        for (int k = 0; k < 5; ++k) {
            float htv[16];
#pragma unroll
            for (int wq = 0; wq < 16; ++wq) {
                float s = 0.0f;
#pragma unroll
                for (int j = 0; j < 5; ++j) s += cf[j][wq] * t[j * 5 + k];
                htv[wq] = PW1_2 * (INV_S5 * s + cres[wv][16 + bl * 5 + k][half * 16 + wq]);
            }
#pragma unroll
            for (int q = 0; q < 8; ++q) {
                HP pk2;
                pk2.h[0] = (_Float16)htv[2 * q]; pk2.h[1] = (_Float16)htv[2 * q + 1];
                *(int*)&htA[wv][bl * 5 + k][half * 16 + 2 * q] = pk2.i;
            }
        }
    }
    asm volatile("s_wait_dscnt 0x0" ::: "memory");

    // ---- tp2 WMMAs: cd = h_s x Wcd (rows 0..15), P = h_t x W2t (rows 16..95) ----
    {
        v16h aH = ldA16(&hstg[wv][m][0], hi8);
#pragma unroll
        for (int nt = 0; nt < 2; ++nt) {
            v8f c = {0.f,0.f,0.f,0.f,0.f,0.f,0.f,0.f};
            c = wmma32(aH, ldB16(&wsm4[2 * 128 + (nt * 32 + lane) * 2]), c);
#pragma unroll
            for (int r = 0; r < 8; ++r) cres[wv][hi8 + r][nt * 16 + m] = c[r];
        }
#pragma unroll
        for (int mt = 0; mt < 5; ++mt) {
            v16h aP = ldA16(&htA[wv][mt * 16 + m][0], hi8);
#pragma unroll
            for (int nt = 0; nt < 2; ++nt) {
                v8f c = {0.f,0.f,0.f,0.f,0.f,0.f,0.f,0.f};
                c = wmma32(aP, ldB16(&wsm4[3 * 128 + (nt * 32 + lane) * 2]), c);
#pragma unroll
                for (int r = 0; r < 8; ++r) cres[wv][16 + mt * 16 + hi8 + r][nt * 16 + m] = c[r];
            }
        }
    }
    asm volatile("s_wait_dscnt 0x0" ::: "memory");

    // ---- epilogue: G[i,j] = sum_v P[i,v] h_t[v,j], term1, Wigner contract ----
    {
        float ht[5][16];                       // h_t[v in half, j]
#pragma unroll
        for (int j = 0; j < 5; ++j) {
            const int4* hp = (const int4*)&htA[wv][bl * 5 + j][half * 16];
            H8 x0, x1; x0.i = hp[0]; x1.i = hp[1];
#pragma unroll
            for (int z = 0; z < 8; ++z) { ht[j][z] = (float)x0.h[z]; ht[j][8 + z] = (float)x1.h[z]; }
        }
        float Gp[25], t1p[5];
        {
            const float4* cdp = (const float4*)&cres[wv][bl][half * 16];
            float cd[16];
#pragma unroll
            for (int q = 0; q < 4; ++q) {
                float4 f = cdp[q];
                cd[q * 4] = f.x; cd[q * 4 + 1] = f.y; cd[q * 4 + 2] = f.z; cd[q * 4 + 3] = f.w;
            }
#pragma unroll
            for (int k = 0; k < 5; ++k) {
                float s = 0.0f;
#pragma unroll
                for (int z = 0; z < 16; ++z) s += cd[z] * ht[k][z];
                t1p[k] = s;
            }
        }
#pragma unroll
        for (int i = 0; i < 5; ++i) {
            const float4* pp = (const float4*)&cres[wv][16 + bl * 5 + i][half * 16];
            float P[16];
#pragma unroll
            for (int q = 0; q < 4; ++q) {
                float4 f = pp[q];
                P[q * 4] = f.x; P[q * 4 + 1] = f.y; P[q * 4 + 2] = f.z; P[q * 4 + 3] = f.w;
            }
#pragma unroll
            for (int j = 0; j < 5; ++j) {
                float s = 0.0f;
#pragma unroll
                for (int z = 0; z < 16; ++z) s += P[z] * ht[j][z];
                Gp[i * 5 + j] = s;
            }
        }
        // combine lane-pair halves via scratch (A1 reused as f32)
        float* scr = (float*)&A1[wv][0][0];
#pragma unroll
        for (int z = 0; z < 25; ++z) scr[bl * 64 + half * 32 + z] = Gp[z];
#pragma unroll
        for (int k = 0; k < 5; ++k) scr[bl * 64 + half * 32 + 25 + k] = t1p[k];
        asm volatile("s_wait_dscnt 0x0" ::: "memory");
        if (half == 0) {
            float G[25];
#pragma unroll
            for (int z = 0; z < 25; ++z) G[z] = scr[bl * 64 + z] + scr[bl * 64 + 32 + z];
            float outv[5];
#pragma unroll
            for (int k = 0; k < 5; ++k) {
                float acc = scr[bl * 64 + 25 + k] + scr[bl * 64 + 32 + 25 + k];
#pragma unroll
                for (int i = 0; i < 5; ++i)
#pragma unroll
                    for (int j = 0; j < 5; ++j)
                        acc += cls[(i * 5 + j) * 5 + k] * G[i * 5 + j];
                outv[k] = PW2_2 * acc;
            }
#pragma unroll
            for (int k = 0; k < 5; ++k) outg[(size_t)(b0 + bl) * 5 + k] = outv[k];
        }
    }
}

// ---------------------------------------------------------------------------
extern "C" void kernel_launch(void* const* d_in, const int* in_sizes, int n_in,
                              void* d_out, int out_size, void* d_ws, size_t ws_size,
                              hipStream_t stream) {
    const float* scalars = (const float*)d_in[0];
    const float* t0 = (const float*)d_in[1];
    const float* t1 = (const float*)d_in[2];
    const float* t2 = (const float*)d_in[3];
    const float* t3 = (const float*)d_in[4];
    const float* t4 = (const float*)d_in[5];
    const float* w1_sss = (const float*)d_in[6];
    const float* w1_stt = (const float*)d_in[7];
    const float* w1_tst = (const float*)d_in[8];
    const float* w1_tts = (const float*)d_in[9];
    const float* w1_ttt = (const float*)d_in[10];
    const float* w2_stt = (const float*)d_in[11];
    const float* w2_tst = (const float*)d_in[12];
    const float* w2_ttt = (const float*)d_in[13];
    float* outg = (float*)d_out;

    // workspace layout (needs ~59 MB)
    char* ws = (char*)d_ws;
    int*      wcat  = (int*)(ws);                                   // 282624 B
    int*      wsm   = (int*)(ws + 282624);                          //   8192 B
    float*    c222  = (float*)(ws + 290816);                        //    512 B
    float*    hs0g  = (float*)(ws + 291328);                        // B*32*4 = 16 MB
    _Float16* coefg = (_Float16*)(ws + 291328 + (size_t)B_TOT*32*4);// B*160*2 = 40 MB

    k_prep_wcat <<<WCAT_DW / 256, 256, 0, stream>>>(w1_sss, w1_stt, w1_tst, wcat);
    k_prep_small<<<8, 256, 0, stream>>>(w1_tts, w1_ttt, w2_stt, w2_tst, w2_ttt, wsm);
    k_prep_c222 <<<1, 32, 0, stream>>>(c222);
    k_stage1    <<<256, 256, 0, stream>>>(scalars, (const int4*)wcat, hs0g, coefg);
    k_stage2    <<<B_TOT / 128, 256, 0, stream>>>(t0, t1, t2, t3, t4,
                                                  hs0g, coefg, (const int4*)wsm, c222, outg);
    (void)in_sizes; (void)n_in; (void)out_size; (void)ws_size;
}
